// _ScaledDotProductAttention_82325933130226
// MI455X (gfx1250) — matrix operand order repaired and checked by reading, not compiled, not run on hardware
//
#include <hip/hip_runtime.h>

typedef __attribute__((ext_vector_type(16))) __bf16 v16bf;
typedef __attribute__((ext_vector_type(8)))  float  v8f;

#define S_LEN  2048
#define D_KDIM 64
#define NEG_BIG (-1.0e9f)
#define SCALE   0.125f   // 1/sqrt(64)

union BF16Bits { unsigned short u; __bf16 h; };

__device__ __forceinline__ __bf16 f2bf(float f) {
    union { float f; unsigned u; } x; x.f = f;
    unsigned r = x.u + 0x7FFFu + ((x.u >> 16) & 1u);   // round-to-nearest-even
    BF16Bits b; b.u = (unsigned short)(r >> 16);
    return b.h;
}

__device__ __forceinline__ void async_wait0() {
#if __has_builtin(__builtin_amdgcn_s_wait_asynccnt)
    __builtin_amdgcn_s_wait_asynccnt(0);
#else
    asm volatile("s_wait_asynccnt 0x0" ::: "memory");
#endif
}

// Async global->LDS 128-bit copy (CDNA5, ASYNCcnt-tracked).
__device__ __forceinline__ void async_copy_b128(unsigned int lds_byte_off,
                                                const void* gptr) {
    unsigned long long ga = (unsigned long long)(size_t)gptr;
    asm volatile("global_load_async_to_lds_b128 %0, %1, off"
                 :: "v"(lds_byte_off), "v"(ga) : "memory");
}

extern "C" __global__ __launch_bounds__(256)
void attn_fwd_cdna5(const float* __restrict__ q,
                    const float* __restrict__ k,   // [B,H,d,S] pre-transposed
                    const float* __restrict__ v,
                    const unsigned char* __restrict__ mask, // [B,1,S,S] bool
                    float* __restrict__ out_ctx,
                    float* __restrict__ out_attn)
{
    __shared__ float         smS[16 * S_LEN];       // 128 KB: scores -> probs
    __shared__ float         smQf[16 * D_KDIM];     // 4 KB: fp32 Q tile
    __shared__ unsigned char smMask[16 * S_LEN];    // 32 KB: mask tile
    __shared__ float         red[16 * 16];
    __shared__ float         rowMax[16];
    __shared__ float         rowInv[16];
    __shared__ float         smPart[8 * 256];       // split-K partials

    const int tid  = threadIdx.x;
    const int lane = tid & 31;
    const int wave = tid >> 5;
    const int half = lane >> 4;                     // hi/lo 16-lane group
    const int l16  = lane & 15;

    const int rtile = blockIdx.x & 127;             // S/16 row tiles
    const int bh    = blockIdx.x >> 7;              // b*H + h
    const int b     = bh >> 4;                      // H == 16
    const int r0    = rtile * 16;

    const float* qb = q + (size_t)bh * S_LEN * D_KDIM;
    const float* kb = k + (size_t)bh * D_KDIM * S_LEN;
    const float* vb = v + (size_t)bh * S_LEN * D_KDIM;
    const unsigned char* mb = mask + (size_t)b * S_LEN * S_LEN;

    // ---- async-stage Q tile (16x64 fp32 = 4 KB, contiguous) ----
    {
        const float* qsrc = qb + (size_t)r0 * D_KDIM;     // 1024 floats
        unsigned int qlds = (unsigned int)(size_t)(&smQf[0]);
        async_copy_b128(qlds + tid * 16, qsrc + tid * 4); // 256 x 16B = 4 KB
        // 1024 floats = 4 KB total; each thread moves one 16B chunk (4 floats)
    }
    // ---- async-stage mask tile (16 full rows = 32 KB, contiguous) ----
    {
        const unsigned char* msrc = mb + (size_t)r0 * S_LEN;
        unsigned int mlds = (unsigned int)(size_t)(&smMask[0]);
        #pragma unroll
        for (int i = 0; i < 8; ++i) {
            int idx = i * 256 + tid;                 // 2048 chunks of 16 B
            async_copy_b128(mlds + idx * 16, msrc + idx * 16);
        }
    }
    async_wait0();
    __syncthreads();

    // ---- A fragments (Q), reused for every column tile.
    // ISA 16-bit A 16x32 layout: lane = M (0..15), elem j -> K:
    //   j<8 : K = half*8 + j ; j>=8 : K = 16 + half*8 + (j-8)
    v16bf aq[2];
    #pragma unroll
    for (int kc = 0; kc < 2; ++kc) {
        #pragma unroll
        for (int j = 0; j < 16; ++j) {
            int kk = (j < 8) ? (half * 8 + j) : (16 + half * 8 + (j - 8));
            aq[kc][j] = f2bf(smQf[l16 * D_KDIM + kc * 32 + kk]);
        }
    }

    // ---- phase 1: S = (Q K) * scale, masked; write into LDS ----
    for (int ctile = wave; ctile < S_LEN / 16; ctile += 8) {
        const int c0 = ctile * 16;
        v8f acc = {};
        #pragma unroll
        for (int kc = 0; kc < 2; ++kc) {
            // B 32x16 layout: elem j<8 -> K = half*8 + j; j>=8 -> K = 16 + half*8 + (j-8); N = lane&15
            v16bf bk;
            #pragma unroll
            for (int j = 0; j < 16; ++j) {
                int kk = (j < 8) ? (half * 8 + j) : (16 + half * 8 + (j - 8));
                bk[j] = f2bf(kb[(size_t)(kc * 32 + kk) * S_LEN + c0 + l16]);
            }
            acc = __builtin_amdgcn_wmma_f32_16x16x32_bf16(
                      false, aq[kc], false, bk, (short)0, acc, false, false);
        }
        // C layout: VGPR v -> M = v + 8*half, N = lane&15
        #pragma unroll
        for (int vv = 0; vv < 8; ++vv) {
            int m = vv + half * 8;
            float s = acc[vv] * SCALE;
            if (smMask[m * S_LEN + c0 + l16]) s = NEG_BIG;
            smS[m * S_LEN + c0 + l16] = s;
        }
    }
    __syncthreads();

    // ---- phase 2: exact softmax over each of the 16 rows ----
    const int srow = tid >> 4;                      // 16 threads per row
    const int st   = tid & 15;
    float mx = -3.4e38f;
    for (int c = st; c < S_LEN; c += 16) mx = fmaxf(mx, smS[srow * S_LEN + c]);
    red[srow * 16 + st] = mx;
    __syncthreads();
    if (st == 0) {
        float m2 = red[srow * 16];
        for (int i = 1; i < 16; ++i) m2 = fmaxf(m2, red[srow * 16 + i]);
        rowMax[srow] = m2;
    }
    __syncthreads();
    const float rm = rowMax[srow];
    float sum = 0.f;
    for (int c = st; c < S_LEN; c += 16) {
        float e = __expf(smS[srow * S_LEN + c] - rm);
        smS[srow * S_LEN + c] = e;
        sum += e;
    }
    red[srow * 16 + st] = sum;
    __syncthreads();
    if (st == 0) {
        float s2 = 0.f;
        for (int i = 0; i < 16; ++i) s2 += red[srow * 16 + i];
        rowInv[srow] = 1.0f / s2;
    }
    __syncthreads();

    // normalize, stream attn to HBM (coalesced), keep probs in LDS
    float* attn_row = out_attn + ((size_t)bh * S_LEN + r0) * S_LEN;
    for (int i = 0; i < 128; ++i) {
        int idx = i * 256 + tid;                    // 16*2048 contiguous
        float p = smS[idx] * rowInv[idx >> 11];
        smS[idx] = p;
        attn_row[idx] = p;
    }
    __syncthreads();

    // ---- phase 3: context = P (16x2048) @ V (2048x64), split-K over wave pairs
    const int ctcol = wave & 3;                     // which 16-col tile of d
    const int khalf = wave >> 2;                    // K half: [0,1024) / [1024,2048)
    v8f accc = {};
    for (int ch = 0; ch < 32; ++ch) {
        const int ks0 = khalf * 1024 + ch * 32;
        v16bf ap, bv;
        #pragma unroll
        for (int j = 0; j < 16; ++j) {
            int kk = (j < 8) ? (half * 8 + j) : (16 + half * 8 + (j - 8));
            ap[j] = f2bf(smS[l16 * S_LEN + ks0 + kk]);
        }
        #pragma unroll
        for (int j = 0; j < 16; ++j) {
            int kk = (j < 8) ? (half * 8 + j) : (16 + half * 8 + (j - 8));
            bv[j] = f2bf(vb[(size_t)(ks0 + kk) * D_KDIM + ctcol * 16 + l16]);
        }
        accc = __builtin_amdgcn_wmma_f32_16x16x32_bf16(
                   false, ap, false, bv, (short)0, accc, false, false);
    }
    #pragma unroll
    for (int vv = 0; vv < 8; ++vv) smPart[wave * 256 + lane * 8 + vv] = accc[vv];
    __syncthreads();
    if (wave < 4) {
        float* ctx = out_ctx + ((size_t)bh * S_LEN + r0) * D_KDIM;
        #pragma unroll
        for (int vv = 0; vv < 8; ++vv) {
            float s2 = smPart[wave * 256 + lane * 8 + vv]
                     + smPart[(wave + 4) * 256 + lane * 8 + vv];
            int m = vv + half * 8;
            ctx[(size_t)m * D_KDIM + ctcol * 16 + l16] = s2;
        }
    }
}

extern "C" void kernel_launch(void* const* d_in, const int* in_sizes, int n_in,
                              void* d_out, int out_size, void* d_ws, size_t ws_size,
                              hipStream_t stream) {
    const float* q = (const float*)d_in[0];
    const float* k = (const float*)d_in[1];
    const float* v = (const float*)d_in[2];
    const unsigned char* mask = (const unsigned char*)d_in[3];

    float* out      = (float*)d_out;
    const size_t ctx_elems = (size_t)4 * 16 * 2048 * 64;
    float* out_ctx  = out;
    float* out_attn = out + ctx_elems;

    dim3 grid(4 * 16 * (2048 / 16));   // 8192 blocks: (b,h,row-tile)
    attn_fwd_cdna5<<<grid, 256, 0, stream>>>(q, k, v, mask, out_ctx, out_attn);
}
